// MyGRUFor4Dims_35493609734547
// MI455X (gfx1250) — matrix operand order repaired and checked by reading, not compiled, not run on hardware
//
#include <hip/hip_runtime.h>

// GRU over T2: x [B=32, T1=128, T2=256, D=64] f32, U=D=64, 4096 independent rows.
// Transposed GEMM: h^T = W^T (A operand, LDS-resident) x g^T (B operand).
// 16 rows per wave (N=16), U=64 -> 4 M-tiles, K=128 -> 4 k-frags, 3 gates.
// x[t+1] tile double-buffered in registers to hide HBM latency.

typedef __attribute__((ext_vector_type(16))) _Float16 v16h;
typedef __attribute__((ext_vector_type(8)))  _Float16 v8h;
typedef __attribute__((ext_vector_type(8)))  float    v8f;
typedef __attribute__((ext_vector_type(4)))  float    f4;

#define T2V 256
#define DV  64
#define UV  64
#define ROWS 4096                 // B*T1
#define WAVES_PER_BLK 4
#define ROWS_PER_WAVE 16
#define ROWS_PER_BLK (WAVES_PER_BLK * ROWS_PER_WAVE)  // 64
#define NBLK (ROWS / ROWS_PER_BLK)                    // 64
#define FRAG_HALFS 512            // 32 lanes * 16 halves per A fragment
#define NFRAG 48                  // 3 gates * 4 m-tiles * 4 k-frags

#define WMMA_F16(A, B, C) \
  __builtin_amdgcn_wmma_f32_16x16x32_f16(false, (A), false, (B), (short)0, (C), false, false)

__device__ __forceinline__ float fast_sigmoid(float x) {
  // v_exp + v_rcp (TRANS pipe) instead of IEEE divide expansion
  return __builtin_amdgcn_rcpf(1.0f + __expf(-x));
}

__device__ __forceinline__ float fast_tanh(float x) {
  // tanh(x) = 1 - 2/(exp(2x)+1)
  return 1.0f - 2.0f * __builtin_amdgcn_rcpf(1.0f + __expf(2.0f * x));
}

__device__ __forceinline__ int swz16(int v) {
  // group-of-32 swizzle: xor_mask=0x10, and_mask=0x1f -> swap lane halves
  return __builtin_amdgcn_ds_swizzle(v, 0x401F);
}

__device__ __forceinline__ v16h lda16(const _Float16* p) {
  v16h v;
#pragma unroll
  for (int i = 0; i < 16; ++i) v[i] = p[i];
  return v;
}

// Issue next-step x loads: 2 x 16 contiguous f32 per lane (4x b128 each half).
__device__ __forceinline__ void load_x(const float* px, int koff, f4* dst) {
  const f4* p0 = (const f4*)(px + koff);
  const f4* p1 = (const f4*)(px + 32 + koff);
#pragma unroll
  for (int i = 0; i < 4; ++i) {
    dst[i]     = p0[i];
    dst[4 + i] = p1[i];
  }
}

// B fragment (K slice of 32) from 4 buffered float4 quads (16 f32).
__device__ __forceinline__ v16h b_from_buf(const f4* q) {
  v16h b;
#pragma unroll
  for (int j = 0; j < 4; ++j)
#pragma unroll
    for (int i = 0; i < 4; ++i)
      b[j * 4 + i] = (_Float16)q[j][i];
  return b;
}

// B fragment (one 32-K slice) from two C-layout f32 accumulators (M-tiles
// covering u..u+15 and u+16..u+31). C layout gives each lane 8 M values; the
// other 8 live in the partner lane (lane^16) -> one ds_swizzle half-swap.
__device__ __forceinline__ v16h b_from_cc(v8f c0, v8f c1, bool lolane) {
  union U8 { v8h h; int d[4]; };
  U8 olo, ohi, slo, shi;
#pragma unroll
  for (int i = 0; i < 8; ++i) {
    olo.h[i] = (_Float16)c0[i];
    ohi.h[i] = (_Float16)c1[i];
  }
#pragma unroll
  for (int i = 0; i < 4; ++i) {
    slo.d[i] = swz16(olo.d[i]);
    shi.d[i] = swz16(ohi.d[i]);
  }
  union U16 { v16h h; int d[8]; } r;
#pragma unroll
  for (int i = 0; i < 4; ++i) {
    r.d[i]     = lolane ? olo.d[i] : shi.d[i];  // K 0..7   | K 16..23
    r.d[4 + i] = lolane ? slo.d[i] : ohi.d[i];  // K 8..15  | K 24..31
  }
  return r.h;
}

// One gate's [16 u, 16 row] output tile: 4 k-frag WMMAs, A = weights from LDS.
__device__ __forceinline__ v8f gate4(const _Float16* wbase, v8f acc,
                                     v16h b0, v16h b1, v16h b2, v16h b3) {
  acc = WMMA_F16(lda16(wbase + 0 * FRAG_HALFS), b0, acc);
  acc = WMMA_F16(lda16(wbase + 1 * FRAG_HALFS), b1, acc);
  acc = WMMA_F16(lda16(wbase + 2 * FRAG_HALFS), b2, acc);
  acc = WMMA_F16(lda16(wbase + 3 * FRAG_HALFS), b3, acc);
  return acc;
}

__global__ __launch_bounds__(ROWS_PER_BLK * 2) void gru4d_wmma_t_kernel(
    const float* __restrict__ x,
    const float* __restrict__ kh, const float* __restrict__ kr,
    const float* __restrict__ kz,
    const float* __restrict__ bh, const float* __restrict__ br,
    const float* __restrict__ bz,
    float* __restrict__ out) {
  __shared__ __align__(32) _Float16 wlds[NFRAG * FRAG_HALFS];  // 48 KB
  __shared__ __align__(32) float    blds[3 * 4 * 32 * 8];      // 12 KB

  const int tid  = threadIdx.x;
  const int lane = tid & 31;
  const int wave = tid >> 5;

  // ---- Stage W^T as A fragments (16-bit A 16x32, contiguous K order k = 16h + s):
  // frag f = (g*4 + mt)*4 + kf. Lane l<16: M(row)=u=mt*16+l, K={kb..kb+15};
  // lane>=16: same u (l&15), K={kb+16..kb+31}.
  // A[u][k] = kernel[k][u], kernel row-major [128, 64].
  for (int e = tid; e < NFRAG * 32; e += blockDim.x) {
    int l  = e & 31;
    int f  = e >> 5;
    int g  = f >> 4;
    int mt = (f >> 2) & 3;
    int kf = f & 3;
    const float* W = (g == 0) ? kh : (g == 1) ? kr : kz;
    int u   = mt * 16 + (l & 15);
    int khf = (l < 16) ? 0 : 16;
    int kb  = kf * 32;
    _Float16* dst = wlds + f * FRAG_HALFS + l * 16;
#pragma unroll
    for (int i = 0; i < 8; ++i) dst[i]     = (_Float16)W[(kb + khf + i) * UV + u];
#pragma unroll
    for (int i = 0; i < 8; ++i) dst[8 + i] = (_Float16)W[(kb + 8 + khf + i) * UV + u];
  }

  // ---- Stage biases in per-lane C-layout (lane<16: u=mt*16+0..7, else +8..15).
  for (int e = tid; e < 3 * 4 * 32 * 8; e += blockDim.x) {
    int i  = e & 7;
    int l  = (e >> 3) & 31;
    int mt = (e >> 8) & 3;
    int g  = e >> 10;
    const float* bsrc = (g == 0) ? bh : (g == 1) ? br : bz;
    blds[e] = bsrc[mt * 16 + ((l < 16) ? 0 : 8) + i];
  }
  __syncthreads();

  const int  lm     = lane & 15;
  const bool lolane = lane < 16;
  const int  mb     = lolane ? 0 : 8;   // C-layout M offset for this lane
  const int  koff   = lolane ? 0 : 16;  // B-fragment K offset for this lane
  const int  wave_row0 = blockIdx.x * ROWS_PER_BLK + wave * ROWS_PER_WAVE;
  const int  brow = wave_row0 + lm;     // this lane's column = row index
  const float* xrow = x + (size_t)brow * T2V * DV;

  // ---- h0 = x[:, :, 0, :]^T in C layout: contiguous 8 floats per m-tile.
  v8f h[4];
#pragma unroll
  for (int mt = 0; mt < 4; ++mt) {
    const float* p = xrow + mt * 16 + mb;
#pragma unroll
    for (int i = 0; i < 8; ++i) h[mt][i] = p[i];
  }

  // ---- Register double buffer for the x[t] tile.
  f4 xbuf[2][8];
  load_x(xrow + DV, koff, xbuf[1]);  // t = 1

#pragma unroll 2
  for (int t = 1; t < T2V; ++t) {
    const f4* cur = xbuf[t & 1];
    f4*       nxt = xbuf[(t & 1) ^ 1];
    if (t + 1 < T2V)
      load_x(xrow + (size_t)(t + 1) * DV, koff, nxt);   // hide HBM latency
    if (t + 4 < T2V)
      __builtin_prefetch(xrow + (size_t)(t + 4) * DV + koff, 0, 1);

    v16h b0  = b_from_buf(cur);            // g features  0..31 (x)
    v16h b1  = b_from_buf(cur + 4);        // g features 32..63 (x)
    v16h bh0 = b_from_cc(h[0], h[1], lolane);  // features 64..95  (h)
    v16h bh1 = b_from_cc(h[2], h[3], lolane);  // features 96..127 (h)

    v8f rg[4], zg[4];
#pragma unroll
    for (int mt = 0; mt < 4; ++mt) {
      v8f ar = *(const v8f*)&blds[((1 * 4 + mt) * 32 + lane) * 8];
      v8f az = *(const v8f*)&blds[((2 * 4 + mt) * 32 + lane) * 8];
      rg[mt] = gate4(wlds + (size_t)((1 * 4 + mt) * 4) * FRAG_HALFS + lane * 16,
                     ar, b0, b1, bh0, bh1);
      zg[mt] = gate4(wlds + (size_t)((2 * 4 + mt) * 4) * FRAG_HALFS + lane * 16,
                     az, b0, b1, bh0, bh1);
#pragma unroll
      for (int i = 0; i < 8; ++i) {
        rg[mt][i] = fast_sigmoid(rg[mt][i]);
        zg[mt][i] = fast_sigmoid(zg[mt][i]);
      }
    }

    // Candidate GEMM uses g2 = [x, h*r]; h*r stays in C layout (elementwise).
    v16h bg0 = b_from_cc(h[0] * rg[0], h[1] * rg[1], lolane);
    v16h bg1 = b_from_cc(h[2] * rg[2], h[3] * rg[3], lolane);

#pragma unroll
    for (int mt = 0; mt < 4; ++mt) {
      v8f ah = *(const v8f*)&blds[((0 * 4 + mt) * 32 + lane) * 8];
      v8f c  = gate4(wlds + (size_t)((0 * 4 + mt) * 4) * FRAG_HALFS + lane * 16,
                     ah, b0, b1, bg0, bg1);
#pragma unroll
      for (int i = 0; i < 8; ++i) {
        float hc = fast_tanh(c[i]);
        h[mt][i] = h[mt][i] + zg[mt][i] * (hc - h[mt][i]);
      }
    }
  }

  // ---- h_final [ROWS, U] f32: contiguous 8-float store per m-tile.
#pragma unroll
  for (int mt = 0; mt < 4; ++mt) {
    float* p = out + (size_t)brow * UV + mt * 16 + mb;
#pragma unroll
    for (int i = 0; i < 8; ++i) p[i] = h[mt][i];
  }
}

extern "C" void kernel_launch(void* const* d_in, const int* in_sizes, int n_in,
                              void* d_out, int out_size, void* d_ws, size_t ws_size,
                              hipStream_t stream) {
  (void)in_sizes; (void)n_in; (void)d_ws; (void)ws_size; (void)out_size;
  const float* x  = (const float*)d_in[0];
  const float* kh = (const float*)d_in[1];
  const float* kr = (const float*)d_in[2];
  const float* kz = (const float*)d_in[3];
  const float* bh = (const float*)d_in[4];
  const float* br = (const float*)d_in[5];
  const float* bz = (const float*)d_in[6];
  float* out = (float*)d_out;

  dim3 grid(NBLK);
  dim3 block(ROWS_PER_BLK * 2);  // 128 threads = 4 wave32
  hipLaunchKernelGGL(gru4d_wmma_t_kernel, grid, block, 0, stream,
                     x, kh, kr, kz, bh, br, bz, out);
}
